// EPdisc_77472620085851
// MI455X (gfx1250) — matrix operand order repaired and checked by reading, not compiled, not run on hardware
//
#include <hip/hip_runtime.h>

// ---------------------------------------------------------------------------
// EP free-phase relaxation on MI455X (gfx1250, wave32, WMMA).
// GEMMs run on v_wmma_f32_16x16x32_bf16 (bf16 in, f32 accumulate).
// Weights packed once per call into WMMA-fragment-ordered bf16 (L2-resident);
// states carried in row-major bf16 between steps. Each wave computes a 32x32
// output block (2 A frags x 2 B frags -> 4 WMMAs/k-iter) to halve B re-reads
// from L2, which is the dominant traffic.
// ---------------------------------------------------------------------------

typedef __attribute__((ext_vector_type(16))) __bf16 v16bf;
typedef __attribute__((ext_vector_type(8)))  __bf16 v8bf;
typedef __attribute__((ext_vector_type(8)))  float  v8f;

static constexpr float EP_DT = 1.0f;

// ---- sizes ----------------------------------------------------------------
static constexpr int MB   = 256;   // batch
static constexpr int S0   = 512;
static constexpr int S1   = 2048;
static constexpr int S2   = 2048;
static constexpr int SIN  = 4096;
static constexpr int TSTEPS = 20;

// ---------------------------------------------------------------------------
// fp32 -> bf16 row-major convert (for `data` and the initial states).
// ---------------------------------------------------------------------------
__global__ void conv_bf16_kernel(const float* __restrict__ in, __bf16* __restrict__ out,
                                 long long n) {
  long long idx = (long long)blockIdx.x * blockDim.x + threadIdx.x;
  if (idx < n) out[idx] = (__bf16)in[idx];
}

// ---------------------------------------------------------------------------
// Pack a logical B matrix (K x N) into WMMA B-fragment order, fp32 -> bf16.
//   B[k][n] = transpose ? W[n*ldw + k] : W[k*ldw + n]
// Tile (ko,no): tileIdx = no*(K/32)+ko; lane L (0..31) holds 16 elements e:
//   k = 32*ko + 16*(L/16) + e ,  n = 16*no + (L%16)
// so each lane reads its whole fragment as one contiguous 32-byte load.
// ---------------------------------------------------------------------------
__global__ void pack_b_kernel(const float* __restrict__ W, __bf16* __restrict__ out,
                              int K, int N, int ldw, int transpose) {
  long long total = (long long)K * N;
  long long idx = (long long)blockIdx.x * blockDim.x + threadIdx.x;
  if (idx >= total) return;
  int e = (int)(idx & 15);
  int L = (int)((idx >> 4) & 31);
  long long tile = idx >> 9;
  int K32 = K >> 5;
  int ko = (int)(tile % K32);
  int no = (int)(tile / K32);
  int k = (ko << 5) + ((L >> 4) << 4) + e;
  int n = (no << 4) + (L & 15);
  float v = transpose ? W[(long long)n * ldw + k] : W[(long long)k * ldw + n];
  out[idx] = (__bf16)v;
}

// ---------------------------------------------------------------------------
// 16x32 A fragment from row-major bf16. Contiguous k order, as in pack_b_kernel (lane = 16h+m):
//   elements 0..7  <- K = 16h .. 16h+7
//   elements 8..15 <- K = 16h+8 .. 16h+15
// Two contiguous 16-byte loads, concat via shufflevector.
// ---------------------------------------------------------------------------
__device__ __forceinline__ v16bf load_a_frag(const __bf16* __restrict__ rowp, int h) {
  v8bf lo = *(const v8bf*)(rowp + 16 * h);
  v8bf hi = *(const v8bf*)(rowp + 16 * h + 8);
  return __builtin_shufflevector(lo, hi, 0, 1, 2, 3, 4, 5, 6, 7,
                                 8, 9, 10, 11, 12, 13, 14, 15);
}

// ---------------------------------------------------------------------------
// Accumulate a 32(M) x 32(N) block: rows {arow0, arow0+16}, N tiles {nt0,nt0+1}.
// 2 A frags x 2 B frags -> 4 independent WMMAs per k-iteration.
// ---------------------------------------------------------------------------
__device__ __forceinline__ void gemm_acc(const __bf16* __restrict__ A,
                                         const __bf16* __restrict__ B,
                                         int K, int arow0, int h, int lane, int nt0,
                                         v8f& acc00, v8f& acc01, v8f& acc10, v8f& acc11) {
  const int K32 = K >> 5;
  const __bf16* rowA0 = A + (long long)arow0 * K;
  const __bf16* rowA1 = A + (long long)(arow0 + 16) * K;
  const __bf16* bp0 = B + ((long long)nt0 * K32) * 512 + lane * 16;
  const __bf16* bp1 = B + ((long long)(nt0 + 1) * K32) * 512 + lane * 16;
  for (int ko = 0; ko < K32; ++ko) {
    v16bf a0 = load_a_frag(rowA0 + ko * 32, h);
    v16bf a1 = load_a_frag(rowA1 + ko * 32, h);
    v16bf b0 = *(const v16bf*)(bp0 + (long long)ko * 512);
    v16bf b1 = *(const v16bf*)(bp1 + (long long)ko * 512);
    acc00 = __builtin_amdgcn_wmma_f32_16x16x32_bf16(false, a0, false, b0, (short)0, acc00, false, false);
    acc01 = __builtin_amdgcn_wmma_f32_16x16x32_bf16(false, a0, false, b1, (short)0, acc01, false, false);
    acc10 = __builtin_amdgcn_wmma_f32_16x16x32_bf16(false, a1, false, b0, (short)0, acc10, false, false);
    acc11 = __builtin_amdgcn_wmma_f32_16x16x32_bf16(false, a1, false, b1, (short)0, acc11, false, false);
  }
}

// ---------------------------------------------------------------------------
// Fused EP step GEMM:
//   X = A1 @ B1 (+ A2 @ B2 if K2>0) (+ bias[n]) (+ Cpre[m][n])
//   Sold != null : snew = sold + DT * (rho(X) - sold)  (rho = clamp 0..1)
//                  -> bf16 to SnewB (next-step operand), fp32 to SnewF if set
//   Sold == null : X -> SnewF (raw pre-activation store, for c2)
// Block: 128 threads = 4 waves; each wave -> 32(M) x 32(N); block -> 32 x 128.
// Grid: (N/128, M/32). No divergence before WMMA: EXEC is all-ones.
// ---------------------------------------------------------------------------
__global__ __launch_bounds__(128) void ep_gemm_kernel(
    const __bf16* __restrict__ A1, const __bf16* __restrict__ B1, int K1,
    const __bf16* __restrict__ A2, const __bf16* __restrict__ B2, int K2,
    const float* __restrict__ bias, const float* __restrict__ Cpre,
    const __bf16* __restrict__ Sold, float* __restrict__ SnewF,
    __bf16* __restrict__ SnewB, int N) {
  const int lane = threadIdx.x & 31;
  const int wave = threadIdx.x >> 5;         // 0..3
  const int h    = lane >> 4;
  const int mr   = lane & 15;
  const int mt0  = blockIdx.y * 2;           // two 16-row tiles
  const int nt0  = blockIdx.x * 8 + wave * 2;
  const int arow0 = mt0 * 16 + mr;

  v8f acc00 = {}, acc01 = {}, acc10 = {}, acc11 = {};
  gemm_acc(A1, B1, K1, arow0, h, lane, nt0, acc00, acc01, acc10, acc11);
  if (K2 > 0) gemm_acc(A2, B2, K2, arow0, h, lane, nt0, acc00, acc01, acc10, acc11);

  // C/D layout: element v of lane (16h+n) is row m = 16*mtile + 8h + v, col n.
#pragma unroll
  for (int mi = 0; mi < 2; ++mi) {
#pragma unroll
    for (int t = 0; t < 2; ++t) {
      const v8f& acc = mi == 0 ? (t == 0 ? acc00 : acc01) : (t == 0 ? acc10 : acc11);
#pragma unroll
      for (int v = 0; v < 8; ++v) {
        const int m = (mt0 + mi) * 16 + 8 * h + v;
        const int n = (nt0 + t) * 16 + mr;
        const long long o = (long long)m * N + n;
        float x = acc[v];
        if (bias) x += bias[n];
        if (Cpre) x += Cpre[o];
        if (Sold) {
          float r  = fminf(fmaxf(x, 0.0f), 1.0f);
          float s  = (float)Sold[o];
          float sn = s + EP_DT * (r - s);
          if (SnewB) SnewB[o] = (__bf16)sn;
          if (SnewF) SnewF[o] = sn;
        } else {
          SnewF[o] = x;
        }
      }
    }
  }
}

// ---------------------------------------------------------------------------
extern "C" void kernel_launch(void* const* d_in, const int* in_sizes, int n_in,
                              void* d_out, int out_size, void* d_ws, size_t ws_size,
                              hipStream_t stream) {
  (void)in_sizes; (void)n_in; (void)out_size; (void)ws_size;

  const float* data = (const float*)d_in[0];
  const float* s0in = (const float*)d_in[1];
  const float* s1in = (const float*)d_in[2];
  const float* s2in = (const float*)d_in[3];
  const float* W0   = (const float*)d_in[4];
  const float* b0   = (const float*)d_in[5];
  const float* W1   = (const float*)d_in[6];
  const float* b1   = (const float*)d_in[7];
  const float* W2   = (const float*)d_in[8];
  const float* b2   = (const float*)d_in[9];

  // ---- workspace carve-up -------------------------------------------------
  char* ws = (char*)d_ws;
  size_t off = 0;
  auto carve = [&](size_t bytes) { void* p = ws + off; off += (bytes + 255) & ~(size_t)255; return p; };

  __bf16* Bg0   = (__bf16*)carve((size_t)S1  * S0 * 2);  // W0^T packed  (K=2048,N=512)
  __bf16* Bg1a  = (__bf16*)carve((size_t)S2  * S1 * 2);  // W1^T packed  (K=2048,N=2048)
  __bf16* Bg1b  = (__bf16*)carve((size_t)S0  * S1 * 2);  // W0   packed  (K=512, N=2048)
  __bf16* Bg2   = (__bf16*)carve((size_t)S1  * S2 * 2);  // W1   packed  (K=2048,N=2048)
  __bf16* Bpre  = (__bf16*)carve((size_t)SIN * S2 * 2);  // W2^T packed  (K=4096,N=2048)
  float*  c2    = (float*)carve((size_t)MB * S2 * 4);    // data@W2^T + b2 (fp32)
  __bf16* dataB = (__bf16*)carve((size_t)MB * SIN * 2);  // data in bf16
  __bf16* s0a   = (__bf16*)carve((size_t)MB * S0 * 2);   // bf16 state double-buffers
  __bf16* s0b   = (__bf16*)carve((size_t)MB * S0 * 2);
  __bf16* s1a   = (__bf16*)carve((size_t)MB * S1 * 2);
  __bf16* s1b   = (__bf16*)carve((size_t)MB * S1 * 2);
  __bf16* s2a   = (__bf16*)carve((size_t)MB * S2 * 2);
  __bf16* s2b   = (__bf16*)carve((size_t)MB * S2 * 2);

  // ---- 1) pack weights to bf16 WMMA-fragment order ------------------------
  auto pack = [&](const float* W, __bf16* outp, int K, int N, int ldw, int tr) {
    long long total = (long long)K * N;
    int blocks = (int)((total + 255) / 256);
    pack_b_kernel<<<blocks, 256, 0, stream>>>(W, outp, K, N, ldw, tr);
  };
  pack(W0, Bg0,  S1,  S0, S1,  1);   // B = W0^T
  pack(W1, Bg1a, S2,  S1, S2,  1);   // B = W1^T
  pack(W0, Bg1b, S0,  S1, S1,  0);   // B = W0
  pack(W1, Bg2,  S1,  S2, S2,  0);   // B = W1
  pack(W2, Bpre, SIN, S2, SIN, 1);   // B = W2^T

  // ---- 2) convert data + initial states to bf16 ---------------------------
  auto conv = [&](const float* in, __bf16* outp, long long n) {
    int blocks = (int)((n + 255) / 256);
    conv_bf16_kernel<<<blocks, 256, 0, stream>>>(in, outp, n);
  };
  conv(data, dataB, (long long)MB * SIN);
  conv(s0in, s0a,   (long long)MB * S0);
  conv(s1in, s1a,   (long long)MB * S1);
  conv(s2in, s2a,   (long long)MB * S2);

  // ---- 3) step-invariant precompute: c2 = data @ W2^T + b2 ----------------
  ep_gemm_kernel<<<dim3(S2 / 128, MB / 32), 128, 0, stream>>>(
      dataB, Bpre, SIN, nullptr, nullptr, 0, b2, nullptr,
      nullptr, c2, nullptr, S2);

  // ---- 4) T relaxation steps (double-buffered bf16 states) ----------------
  float* outp = (float*)d_out;
  float* o0f = outp;
  float* o1f = outp + (size_t)MB * S0;
  float* o2f = outp + (size_t)MB * S0 + (size_t)MB * S1;

  __bf16 *cur0 = s0a, *cur1 = s1a, *cur2 = s2a;
  __bf16 *alt0 = s0b, *alt1 = s1b, *alt2 = s2b;

  for (int t = 0; t < TSTEPS; ++t) {
    const bool last = (t == TSTEPS - 1);
    float* f0 = last ? o0f : nullptr;
    float* f1 = last ? o1f : nullptr;
    float* f2 = last ? o2f : nullptr;

    // s0' = s0 + DT*(-s0 + rho(s1 @ W0^T + b0))
    ep_gemm_kernel<<<dim3(S0 / 128, MB / 32), 128, 0, stream>>>(
        cur1, Bg0, S1, nullptr, nullptr, 0, b0, nullptr, cur0, f0, alt0, S0);
    // s1' = s1 + DT*(-s1 + rho(s2 @ W1^T + b1 + s0 @ W0))
    ep_gemm_kernel<<<dim3(S1 / 128, MB / 32), 128, 0, stream>>>(
        cur2, Bg1a, S2, cur0, Bg1b, S0, b1, nullptr, cur1, f1, alt1, S1);
    // s2' = s2 + DT*(-s2 + rho(c2 + s1 @ W1))
    ep_gemm_kernel<<<dim3(S2 / 128, MB / 32), 128, 0, stream>>>(
        cur1, Bg2, S1, nullptr, nullptr, 0, nullptr, c2, cur2, f2, alt2, S2);

    __bf16* tmp;
    tmp = cur0; cur0 = alt0; alt0 = tmp;
    tmp = cur1; cur1 = alt1; alt1 = tmp;
    tmp = cur2; cur2 = alt2; alt2 = tmp;
  }
}
